// MultiHeadYatAttention_50019189129633
// MI455X (gfx1250) — matrix operand order repaired and checked by reading, not compiled, not run on hardware
//
#include <hip/hip_runtime.h>
#include <hip/hip_bf16.h>

typedef __attribute__((ext_vector_type(16))) _Float16 v16h;
typedef __attribute__((ext_vector_type(8)))  _Float16 v8h;
typedef __attribute__((ext_vector_type(8)))  float    v8f;
typedef __attribute__((ext_vector_type(4)))  float    v4f;
typedef __attribute__((ext_vector_type(4)))  int      v4i;

#define AS1 __attribute__((address_space(1)))
#define AS3 __attribute__((address_space(3)))

#if __has_builtin(__builtin_amdgcn_global_load_async_to_lds_b128) && \
    __has_builtin(__builtin_amdgcn_s_wait_asynccnt)
#define HAVE_ASYNC_LDS 1
#else
#define HAVE_ASYNC_LDS 0
#endif

// 16-byte global -> LDS copy; async DMA (ASYNCcnt) when available.
static __device__ __forceinline__ void cp16(_Float16* dst, const _Float16* src) {
#if HAVE_ASYNC_LDS
  __builtin_amdgcn_global_load_async_to_lds_b128((AS1 v4i*)src, (AS3 v4i*)dst,
                                                 0, 0);
#else
  *(v8h*)dst = *(const v8h*)src;
#endif
}

static __device__ __forceinline__ void async_wait_all() {
#if HAVE_ASYNC_LDS
  __builtin_amdgcn_s_wait_asynccnt(0);
#endif
}

static __device__ __forceinline__ v16h combine16(v8h lo, v8h hi) {
  v16h r;
#pragma unroll
  for (int i = 0; i < 8; ++i) { r[i] = lo[i]; r[i + 8] = hi[i]; }
  return r;
}

static __device__ __forceinline__ v8f zero8() {
  v8f z;
#pragma unroll
  for (int i = 0; i < 8; ++i) z[i] = 0.0f;
  return z;
}

static __device__ __forceinline__ v8f wmma_f16(v16h a, v16h b, v8f c) {
  return __builtin_amdgcn_wmma_f32_16x16x32_f16(
      /*neg_a=*/false, a, /*neg_b=*/false, b,
      /*c_mod=*/(short)0, c, /*reuse_a=*/false, /*reuse_b=*/false);
}

// ---------------------------------------------------------------------------
// One-shot f32 -> f16 conversion (bandwidth-bound; keeps cvt out of GEMMs).
// ---------------------------------------------------------------------------
__global__ __launch_bounds__(256) void cvt_f32_f16(
    const float* __restrict__ x, _Float16* __restrict__ y, int n) {
  const int i = (blockIdx.x * blockDim.x + threadIdx.x) * 8;
  if (i >= n) return;
  v4f a = *(const v4f*)(x + i);
  v4f b = *(const v4f*)(x + i + 4);
  v8h r;
#pragma unroll
  for (int t = 0; t < 4; ++t) {
    r[t] = (_Float16)a[t];
    r[t + 4] = (_Float16)b[t];
  }
  *(v8h*)(y + i) = r;
}

// ---------------------------------------------------------------------------
// NT-GEMM (f16 in): C[m,n] = sum_k A[m,k] * W[n,k] + bias[n]
// 256 thr = 8 waves; block tile 128x128, BK=64; wave tile 32x64.
// Double-buffered LDS fed by async global->LDS DMA.
// ---------------------------------------------------------------------------
template <bool OUT_F32>
__global__ __launch_bounds__(256) void gemm_nt_wmma(
    const _Float16* __restrict__ A, const _Float16* __restrict__ W,
    const float* __restrict__ bias, void* __restrict__ Cp,
    int M, int N, int K) {
  constexpr int BK = 64;
  __shared__ _Float16 As[2][128][72];  // row stride 144B (16B multiple)
  __shared__ _Float16 Bs[2][128][72];

  const int tid = threadIdx.x;
  const int lane = tid & 31, wid = tid >> 5;
  const int waveM = wid & 3, waveN = wid >> 2;  // 4 x 2 wave grid
  const int m0 = blockIdx.y * 128, n0 = blockIdx.x * 128;

  v8f acc[2][4];
#pragma unroll
  for (int mi = 0; mi < 2; ++mi)
#pragma unroll
    for (int ni = 0; ni < 4; ++ni) acc[mi][ni] = zero8();

  const int srow = tid >> 1;        // 0..127
  const int sseg = (tid & 1) * 32;  // 0 or 32 (halves)

  auto stage = [&](int kt, int buf) {
    const _Float16* ap = A + (size_t)(m0 + srow) * K + kt + sseg;
    const _Float16* wp = W + (size_t)(n0 + srow) * K + kt + sseg;
#pragma unroll
    for (int c = 0; c < 4; ++c) cp16(&As[buf][srow][sseg + c * 8], ap + c * 8);
#pragma unroll
    for (int c = 0; c < 4; ++c) cp16(&Bs[buf][srow][sseg + c * 8], wp + c * 8);
  };

  const int nk = K / BK;
  stage(0, 0);
  async_wait_all();
  __syncthreads();

  const int ao = (lane < 16) ? 0 : 8;   // A layout: k {0..7,16..23} / {8..15,24..31}
  const int bo = (lane < 16) ? 0 : 8;   // B layout: k {0..7,16..23} / {8..15,24..31}

  for (int t = 0; t < nk; ++t) {
    const int cur = t & 1;
    if (t + 1 < nk) stage((t + 1) * BK, (t + 1) & 1);  // DMA next tile

#pragma unroll
    for (int kk = 0; kk < 2; ++kk) {  // two 32-wide k-steps
      v16h afr[2], bfr[4];
#pragma unroll
      for (int mi = 0; mi < 2; ++mi) {
        const _Float16* p = &As[cur][waveM * 32 + mi * 16 + (lane & 15)][kk * 32 + ao];
        afr[mi] = combine16(*(const v8h*)p, *(const v8h*)(p + 16));
      }
#pragma unroll
      for (int ni = 0; ni < 4; ++ni) {
        const _Float16* p = &Bs[cur][waveN * 64 + ni * 16 + (lane & 15)][kk * 32 + bo];
        bfr[ni] = combine16(*(const v8h*)p, *(const v8h*)(p + 16));
      }
#pragma unroll
      for (int mi = 0; mi < 2; ++mi)
#pragma unroll
        for (int ni = 0; ni < 4; ++ni)
          acc[mi][ni] = wmma_f16(afr[mi], bfr[ni], acc[mi][ni]);
    }

    async_wait_all();
    __syncthreads();
  }

  // epilogue: C layout row = r (+8 for upper half-wave), col = lane&15
#pragma unroll
  for (int mi = 0; mi < 2; ++mi)
#pragma unroll
    for (int ni = 0; ni < 4; ++ni) {
      const int col = n0 + waveN * 64 + ni * 16 + (lane & 15);
      const float bv = bias[col];
#pragma unroll
      for (int r = 0; r < 8; ++r) {
        const int row = m0 + waveM * 32 + mi * 16 + r + ((lane < 16) ? 0 : 8);
        const float val = acc[mi][ni][r] + bv;
        if constexpr (OUT_F32)
          ((float*)Cp)[(size_t)row * N + col] = val;
        else
          ((_Float16*)Cp)[(size_t)row * N + col] = (_Float16)val;
      }
    }
}

// ---------------------------------------------------------------------------
// Per-(b,s,h) squared norms of 64-wide head rows, from f16 [B*S, E] layout.
// ---------------------------------------------------------------------------
__global__ __launch_bounds__(256) void row_sumsq(
    const _Float16* __restrict__ x, float* __restrict__ out, int n) {
  const int i = blockIdx.x * blockDim.x + threadIdx.x;
  if (i >= n) return;
  const _Float16* p = x + (size_t)(i >> 4) * 1024 + (i & 15) * 64;
  float s = 0.0f;
#pragma unroll
  for (int j = 0; j < 64; j += 8) {
    v8h v = *(const v8h*)(p + j);
#pragma unroll
    for (int t = 0; t < 8; ++t) { float f = (float)v[t]; s += f * f; }
  }
  out[i] = s;
}

// ---------------------------------------------------------------------------
// YAT flash attention. Grid: (S/128, B*H). 8 waves x 16 query rows; streams
// 64-key K/V tiles through LDS with online softmax.
// score = (q.k)^2 / (|q|^2+|k|^2-2 q.k + eps) * (sqrt(D)/log1p(D))^alpha
// ---------------------------------------------------------------------------
__global__ __launch_bounds__(256) void yat_flash_attn(
    const _Float16* __restrict__ Q, const _Float16* __restrict__ Kx,
    const _Float16* __restrict__ V, const float* __restrict__ sq,
    const float* __restrict__ sk, const float* __restrict__ alpha,
    _Float16* __restrict__ O) {
  constexpr int S = 2048, H = 16, D = 64, E = 1024;
  __shared__ _Float16 Ks[64][72];     // K tile, row-major [key][d]
  __shared__ _Float16 Vt[64][72];     // V tile transposed [d][key]
  __shared__ float sks[64];
  __shared__ _Float16 Ps[8][16][72];  // per-wave P staging (C->A relayout)

  const int tid = threadIdx.x, lane = tid & 31, wid = tid >> 5;
  const int bh = blockIdx.y, b = bh >> 4, h = bh & 15;
  const int q0 = blockIdx.x * 128 + wid * 16;
  const size_t baseBS = (size_t)b * S;

  const float scale = powf(8.0f / logf(65.0f), alpha[0]);  // sqrt(64)/log1p(64)

  const int ao = (lane < 16) ? 0 : 8;
  const _Float16* qrow = Q + (baseBS + q0 + (lane & 15)) * E + h * D;
  v16h qf[2];
#pragma unroll
  for (int kk = 0; kk < 2; ++kk)
    qf[kk] = combine16(*(const v8h*)(qrow + kk * 32 + ao),
                       *(const v8h*)(qrow + kk * 32 + ao + 16));

  float sqr[8];
#pragma unroll
  for (int r = 0; r < 8; ++r)
    sqr[r] = sq[(baseBS + q0 + r + ((lane < 16) ? 0 : 8)) * H + h];

  float mr[8], lr[8];
  v8f oacc[4];
#pragma unroll
  for (int r = 0; r < 8; ++r) { mr[r] = -1e30f; lr[r] = 0.0f; }
#pragma unroll
  for (int t = 0; t < 4; ++t) oacc[t] = zero8();

  const int srow = tid >> 2;        // 0..63
  const int sseg = (tid & 3) * 16;  // 0,16,32,48
  const int bo = (lane < 16) ? 0 : 8;

  for (int kt = 0; kt < S; kt += 64) {
    // ---- stage K (async DMA), V^T (register transpose), |k|^2 ----
    const _Float16* kp = Kx + (baseBS + kt + srow) * E + h * D + sseg;
    cp16(&Ks[srow][sseg], kp);
    cp16(&Ks[srow][sseg + 8], kp + 8);
    const _Float16* vp = V + (baseBS + kt + srow) * E + h * D + sseg;
    v8h va = *(const v8h*)vp, vb = *(const v8h*)(vp + 8);
#pragma unroll
    for (int i = 0; i < 8; ++i) {
      Vt[sseg + i][srow] = va[i];
      Vt[sseg + 8 + i][srow] = vb[i];
    }
    if (tid < 64) sks[tid] = sk[(baseBS + kt + tid) * H + h];
    async_wait_all();
    __syncthreads();

    // ---- S = Q K^T ----
    v8f sa[4];
#pragma unroll
    for (int j = 0; j < 4; ++j) {
      sa[j] = zero8();
      const _Float16* kb = &Ks[j * 16 + (lane & 15)][0];
#pragma unroll
      for (int kk = 0; kk < 2; ++kk) {
        v16h bf = combine16(*(const v8h*)(kb + kk * 32 + bo),
                            *(const v8h*)(kb + kk * 32 + bo + 16));
        sa[j] = wmma_f16(qf[kk], bf, sa[j]);
      }
    }

    // ---- YAT scores ----
    float sc[4][8];
#pragma unroll
    for (int j = 0; j < 4; ++j) {
      const float skv = sks[j * 16 + (lane & 15)];
#pragma unroll
      for (int r = 0; r < 8; ++r) {
        const float qk = sa[j][r];
        const float dist = sqr[r] + skv - 2.0f * qk + 1e-5f;
        sc[j][r] = (qk * qk / dist) * scale;
      }
    }

    // ---- online softmax ----
    float corr[8];
#pragma unroll
    for (int r = 0; r < 8; ++r) {
      float mt = fmaxf(fmaxf(sc[0][r], sc[1][r]), fmaxf(sc[2][r], sc[3][r]));
#pragma unroll
      for (int off = 1; off < 16; off <<= 1)
        mt = fmaxf(mt, __shfl_xor(mt, off, 32));
      const float mn = fmaxf(mr[r], mt);
      corr[r] = __expf(mr[r] - mn);
      mr[r] = mn;
      lr[r] *= corr[r];
    }
#pragma unroll
    for (int t = 0; t < 4; ++t)
#pragma unroll
      for (int r = 0; r < 8; ++r) oacc[t][r] *= corr[r];

    float ps[8];
#pragma unroll
    for (int r = 0; r < 8; ++r) ps[r] = 0.0f;
#pragma unroll
    for (int j = 0; j < 4; ++j)
#pragma unroll
      for (int r = 0; r < 8; ++r) {
        const float p = __expf(sc[j][r] - mr[r]);
        ps[r] += p;
        Ps[wid][r + ((lane < 16) ? 0 : 8)][j * 16 + (lane & 15)] = (_Float16)p;
      }
#pragma unroll
    for (int r = 0; r < 8; ++r) {
#pragma unroll
      for (int off = 1; off < 16; off <<= 1)
        ps[r] += __shfl_xor(ps[r], off, 32);
      lr[r] += ps[r];
    }
    __syncthreads();

    // ---- O += P V ----
    {
      const _Float16* pr = &Ps[wid][lane & 15][0];
      v16h pf[2];
#pragma unroll
      for (int kk = 0; kk < 2; ++kk)
        pf[kk] = combine16(*(const v8h*)(pr + kk * 32 + ao),
                           *(const v8h*)(pr + kk * 32 + ao + 16));
#pragma unroll
      for (int jd = 0; jd < 4; ++jd) {
        const _Float16* vr = &Vt[jd * 16 + (lane & 15)][0];
#pragma unroll
        for (int kk = 0; kk < 2; ++kk) {
          v16h vf = combine16(*(const v8h*)(vr + kk * 32 + bo),
                              *(const v8h*)(vr + kk * 32 + bo + 16));
          oacc[jd] = wmma_f16(pf[kk], vf, oacc[jd]);
        }
      }
    }
    __syncthreads();
  }

  // ---- normalize + write f16 attention output [B*S, E] ----
#pragma unroll
  for (int jd = 0; jd < 4; ++jd) {
    const int col = h * D + jd * 16 + (lane & 15);
#pragma unroll
    for (int r = 0; r < 8; ++r) {
      const int row = q0 + r + ((lane < 16) ? 0 : 8);
      O[(baseBS + row) * E + col] = (_Float16)(oacc[jd][r] / lr[r]);
    }
  }
}

// ---------------------------------------------------------------------------
extern "C" void kernel_launch(void* const* d_in, const int* in_sizes, int n_in,
                              void* d_out, int out_size, void* d_ws, size_t ws_size,
                              hipStream_t stream) {
  (void)in_sizes; (void)n_in; (void)out_size; (void)ws_size;
  constexpr int Bc = 2, S = 2048, E = 1024, H = 16;
  constexpr int M = Bc * S;              // 4096
  constexpr size_t NE = (size_t)M * E;   // 4M elems
  constexpr size_t EE = (size_t)E * E;   // 1M elems

  const float* query = (const float*)d_in[0];
  const float* Wq = (const float*)d_in[1];
  const float* bq = (const float*)d_in[2];
  const float* Wk = (const float*)d_in[3];
  const float* bk = (const float*)d_in[4];
  const float* Wv = (const float*)d_in[5];
  const float* bv = (const float*)d_in[6];
  const float* Wo = (const float*)d_in[7];
  const float* bo = (const float*)d_in[8];
  const float* alpha = (const float*)d_in[9];

  _Float16* qh = (_Float16*)d_ws;
  _Float16* kh = qh + NE;
  _Float16* vh = kh + NE;
  _Float16* oh = vh + NE;
  _Float16* xh = oh + NE;        // query in f16
  _Float16* wqh = xh + NE;
  _Float16* wkh = wqh + EE;
  _Float16* wvh = wkh + EE;
  _Float16* woh = wvh + EE;
  float* sqv = (float*)(woh + EE);
  float* skv = sqv + (size_t)M * H;

  // f32 -> f16 staging (activations + weights)
  cvt_f32_f16<<<(int)(NE / 8 / 256), 256, 0, stream>>>(query, xh, (int)NE);
  cvt_f32_f16<<<(int)(EE / 8 / 256), 256, 0, stream>>>(Wq, wqh, (int)EE);
  cvt_f32_f16<<<(int)(EE / 8 / 256), 256, 0, stream>>>(Wk, wkh, (int)EE);
  cvt_f32_f16<<<(int)(EE / 8 / 256), 256, 0, stream>>>(Wv, wvh, (int)EE);
  cvt_f32_f16<<<(int)(EE / 8 / 256), 256, 0, stream>>>(Wo, woh, (int)EE);

  dim3 gg(E / 128, M / 128);  // (8, 32)
  gemm_nt_wmma<false><<<gg, 256, 0, stream>>>(xh, wqh, bq, qh, M, E, E);
  gemm_nt_wmma<false><<<gg, 256, 0, stream>>>(xh, wkh, bk, kh, M, E, E);
  gemm_nt_wmma<false><<<gg, 256, 0, stream>>>(xh, wvh, bv, vh, M, E, E);

  const int nrows = M * H;  // 65536
  row_sumsq<<<(nrows + 255) / 256, 256, 0, stream>>>(qh, sqv, nrows);
  row_sumsq<<<(nrows + 255) / 256, 256, 0, stream>>>(kh, skv, nrows);

  yat_flash_attn<<<dim3(S / 128, Bc * H), 256, 0, stream>>>(qh, kh, vh, sqv, skv,
                                                            alpha, oh);

  gemm_nt_wmma<true><<<gg, 256, 0, stream>>>(oh, woh, bo, (float*)d_out, M, E, E);
}
